// OwlWrapper_82978768159630
// MI455X (gfx1250) — matrix operand order repaired and checked by reading, not compiled, not run on hardware
//
#include <hip/hip_runtime.h>
#include <hip/hip_bf16.h>

// ---------------------------------------------------------------------------
// OWLv2 ViT forward for MI455X (gfx1250), wave32 + WMMA f16->f32.
// All GEMMs run through one tiled v_wmma_f32_16x16x32_f16 kernel with
// double-buffered LDS staging. Staging uses the CDNA5 async direct
// global->LDS path (ASYNCcnt) when the builtin is available.
// ---------------------------------------------------------------------------

typedef _Float16 half_t;
typedef __attribute__((ext_vector_type(16))) _Float16 v16h;
typedef __attribute__((ext_vector_type(8)))  _Float16 v8h;
typedef __attribute__((ext_vector_type(8)))  float    v8f;
typedef int v4i_t __attribute__((vector_size(16)));

#define BM 128
#define BN 128
#define BK 32
#define SK 40   // padded LDS k-stride (halfs); 80B row pitch, 16B aligned

// CDNA5 async global->LDS staging (tracked by ASYNCcnt), if toolchain exposes it
#if defined(__HIP_DEVICE_COMPILE__) && defined(__gfx1250__) && \
    __has_builtin(__builtin_amdgcn_global_load_async_to_lds_b128)
#define ASYNC_STAGE 1
#else
#define ASYNC_STAGE 0
#endif

// Address-space bridging via integer casts: flat shared pointers carry the
// LDS byte offset in their low 32 bits; global flat addrs pass through.
#define TO_LDS_V4(p) \
    ((__attribute__((address_space(3))) v4i_t*)(unsigned)(unsigned long long)(p))
#define TO_GLB_V4(p) \
    ((__attribute__((address_space(1))) v4i_t*)(unsigned long long)(p))

__device__ __forceinline__ void async_wait0()
{
#if ASYNC_STAGE
#if __has_builtin(__builtin_amdgcn_s_wait_asynccnt)
    __builtin_amdgcn_s_wait_asynccnt(0);
#else
    asm volatile("s_wait_asynccnt 0x0" ::: "memory");
#endif
#endif
}

// ===========================================================================
// Generic WMMA GEMM:  C[M,N] = act(alpha*(A[M,K] x B + bias)) (+ residual)
//   A: f16 row-major, leading dim lda (K <= lda, K % 32 == 0)
//   B: bKN==0 -> f16 [N,K] row-major (weight layout, x @ W^T)
//      bKN==1 -> f16 [K,N] row-major (e.g. V in attention)
//   Kv: valid K extent of B (rows beyond Kv read as zero)
//   act: 0 none, 1 quick_gelu, 2 exact gelu
// ===========================================================================
__global__ __launch_bounds__(256)
void wmma_gemm_f16(const half_t* __restrict__ A, int lda,
                   const half_t* __restrict__ B, int ldb, int bKN,
                   const float* __restrict__ bias,
                   const float* __restrict__ residual, int ldr,
                   float* __restrict__ outF, half_t* __restrict__ outH, int ldc,
                   int M, int N, int K, int Kv,
                   float alpha, int act)
{
    __shared__ half_t As[2][BM * SK];
    __shared__ half_t Bs[2][BN * SK];

    const int tid  = threadIdx.x;
    const int lane = tid & 31;
    const int wid  = tid >> 5;      // 0..7
    const int wm   = wid >> 2;      // 0..1  (wave rows)
    const int wn   = wid & 3;       // 0..3  (wave cols)
    const int m0   = blockIdx.y * BM;
    const int n0   = blockIdx.x * BN;

    v8f acc[4][2];
#pragma unroll
    for (int tm = 0; tm < 4; ++tm)
#pragma unroll
        for (int tn = 0; tn < 2; ++tn)
#pragma unroll
            for (int r = 0; r < 8; ++r) acc[tm][tn][r] = 0.f;

#if ASYNC_STAGE
    // Zero-prefill both LDS buffers once: lanes whose tile element is OOB
    // never issue an async copy, so those slots must already hold zero.
    {
        half_t* fa = &As[0][0];
        half_t* fb = &Bs[0][0];
        v8h z;
#pragma unroll
        for (int j = 0; j < 8; ++j) z[j] = (half_t)0.f;
        for (int i = tid; i < 2 * BM * SK / 8; i += 256) {
            *reinterpret_cast<v8h*>(fa + i * 8) = z;
            *reinterpret_cast<v8h*>(fb + i * 8) = z;
        }
        __syncthreads();
    }
#endif

    auto stageA = [&](int buf, int kb) {
#pragma unroll
        for (int it = 0; it < 2; ++it) {
            int cid = tid + it * 256;
            int r   = cid >> 2;
            int ko  = (cid & 3) * 8;
            int gm  = m0 + r;
#if ASYNC_STAGE
            if (gm < M)
                __builtin_amdgcn_global_load_async_to_lds_b128(
                    TO_GLB_V4(A + (size_t)gm * lda + kb + ko),
                    TO_LDS_V4(&As[buf][r * SK + ko]), 0, 0);
#else
            v8h v;
#pragma unroll
            for (int j = 0; j < 8; ++j) v[j] = (half_t)0.f;
            if (gm < M)
                v = *reinterpret_cast<const v8h*>(A + (size_t)gm * lda + kb + ko);
            *reinterpret_cast<v8h*>(&As[buf][r * SK + ko]) = v;
#endif
        }
    };

    auto stageB = [&](int buf, int kb) {
        if (bKN == 0) {  // B is [N,K] row-major, contiguous along k
#pragma unroll
            for (int it = 0; it < 2; ++it) {
                int cid = tid + it * 256;
                int r   = cid >> 2;
                int ko  = (cid & 3) * 8;
                int gn  = n0 + r;
#if ASYNC_STAGE
                if (gn < N && kb + ko + 8 <= Kv) {
                    __builtin_amdgcn_global_load_async_to_lds_b128(
                        TO_GLB_V4(B + (size_t)gn * ldb + kb + ko),
                        TO_LDS_V4(&Bs[buf][r * SK + ko]), 0, 0);
                } else if (gn < N) {
                    v8h v;
#pragma unroll
                    for (int j = 0; j < 8; ++j) v[j] = (half_t)0.f;
                    for (int j = 0; j < 8; ++j)
                        if (kb + ko + j < Kv)
                            v[j] = B[(size_t)gn * ldb + kb + ko + j];
                    *reinterpret_cast<v8h*>(&Bs[buf][r * SK + ko]) = v;
                }
#else
                v8h v;
#pragma unroll
                for (int j = 0; j < 8; ++j) v[j] = (half_t)0.f;
                if (gn < N) {
                    if (kb + ko + 8 <= Kv) {
                        v = *reinterpret_cast<const v8h*>(B + (size_t)gn * ldb + kb + ko);
                    } else {
                        for (int j = 0; j < 8; ++j)
                            if (kb + ko + j < Kv)
                                v[j] = B[(size_t)gn * ldb + kb + ko + j];
                    }
                }
                *reinterpret_cast<v8h*>(&Bs[buf][r * SK + ko]) = v;
#endif
            }
        } else {         // B is [K,N] row-major: stage transposed into LDS (sync)
#pragma unroll
            for (int it = 0; it < 2; ++it) {
                int cid = tid + it * 256;
                int k   = cid >> 4;        // 0..31
                int n   = (cid & 15) * 8;  // 0..120
                int gk  = kb + k;
#pragma unroll
                for (int j = 0; j < 8; ++j) {
                    half_t v = (half_t)0.f;
                    if (gk < Kv && (n0 + n + j) < N)
                        v = B[(size_t)gk * ldb + n0 + n + j];
                    Bs[buf][(n + j) * SK + k] = v;
                }
            }
        }
    };

    stageA(0, 0);
    stageB(0, 0);
    async_wait0();
    __syncthreads();

    const int nk  = K / BK;
    const int hi  = lane >> 4;
    const int l15 = lane & 15;

    for (int s = 0; s < nk; ++s) {
        int cur = s & 1;
        if (s + 1 < nk) { stageA(cur ^ 1, (s + 1) * BK); stageB(cur ^ 1, (s + 1) * BK); }

        // A fragments: 16x32 f16. lane m=(lane&15), hi selects k-halves:
        //   halfs 0..7 -> k = 8*hi .. 8*hi+7 ; halfs 8..15 -> k = 16+8*hi ..
        v16h af[4];
#pragma unroll
        for (int tm = 0; tm < 4; ++tm) {
            int m = wm * 64 + tm * 16 + l15;
            v8h lo = *reinterpret_cast<const v8h*>(&As[cur][m * SK + 8 * hi]);
            v8h hh = *reinterpret_cast<const v8h*>(&As[cur][m * SK + 16 + 8 * hi]);
            af[tm] = __builtin_shufflevector(lo, hh, 0,1,2,3,4,5,6,7,8,9,10,11,12,13,14,15);
        }
        // B fragments: 32x16 f16. lane n=(lane&15), hi selects k-halves:
        //   halfs 0..7 -> k = 8*hi .. 8*hi+7 ; halfs 8..15 -> k = 16+8*hi ..
        v16h bf[2];
#pragma unroll
        for (int tn = 0; tn < 2; ++tn) {
            int n = wn * 32 + tn * 16 + l15;
            v8h lo = *reinterpret_cast<const v8h*>(&Bs[cur][n * SK + 8 * hi]);
            v8h hh = *reinterpret_cast<const v8h*>(&Bs[cur][n * SK + 16 + 8 * hi]);
            bf[tn] = __builtin_shufflevector(lo, hh, 0,1,2,3,4,5,6,7,8,9,10,11,12,13,14,15);
        }

#pragma unroll
        for (int tm = 0; tm < 4; ++tm)
#pragma unroll
            for (int tn = 0; tn < 2; ++tn)
                acc[tm][tn] = __builtin_amdgcn_wmma_f32_16x16x32_f16(
                    false, af[tm], false, bf[tn], (short)0, acc[tm][tn], false, false);

        async_wait0();
        __syncthreads();
    }

    // Epilogue. D layout: VGPR r -> M = r + 8*hi, N = lane&15.
    for (int tm = 0; tm < 4; ++tm) {
        for (int tn = 0; tn < 2; ++tn) {
            int n = n0 + wn * 32 + tn * 16 + l15;
            if (n >= N) continue;
            float bv = bias ? bias[n] : 0.f;
#pragma unroll
            for (int r = 0; r < 8; ++r) {
                int m = m0 + wm * 64 + tm * 16 + hi * 8 + r;
                if (m >= M) continue;
                float x = (acc[tm][tn][r] + bv) * alpha;
                if (act == 1)      x = x / (1.f + __expf(-1.702f * x));
                else if (act == 2) x = 0.5f * x * (1.f + erff(x * 0.70710678118654752f));
                if (residual) x += residual[(size_t)m * ldr + n];
                if (outF) outF[(size_t)m * ldc + n] = x;
                if (outH) outH[(size_t)m * ldc + n] = (half_t)x;
            }
        }
    }
}

// ===========================================================================
// Supporting kernels
// ===========================================================================
__global__ __launch_bounds__(256)
void layernorm_k(const float* __restrict__ in, int ldi,
                 const float* __restrict__ g, const float* __restrict__ b,
                 float* __restrict__ o32, half_t* __restrict__ o16, int ldo, int D)
{
    __shared__ float r1[256], r2[256];
    int row = blockIdx.x, tid = threadIdx.x;
    const float* x = in + (size_t)row * ldi;
    float s = 0.f, ss = 0.f;
    for (int c = tid; c < D; c += 256) { float v = x[c]; s += v; ss += v * v; }
    r1[tid] = s; r2[tid] = ss; __syncthreads();
    for (int w = 128; w > 0; w >>= 1) {
        if (tid < w) { r1[tid] += r1[tid + w]; r2[tid] += r2[tid + w]; }
        __syncthreads();
    }
    float mean = r1[0] / D;
    float var  = r2[0] / D - mean * mean;
    float inv  = rsqrtf(var + 1e-5f);
    for (int c = tid; c < D; c += 256) {
        float y = (x[c] - mean) * inv * g[c] + b[c];
        if (o32) o32[(size_t)row * ldo + c] = y;
        if (o16) o16[(size_t)row * ldo + c] = (half_t)y;
    }
}

__global__ __launch_bounds__(256)
void softmax_f16_k(const float* __restrict__ S, int lds,
                   half_t* __restrict__ P, int ldp, int Nv, int Np)
{
    __shared__ float red[256];
    int row = blockIdx.x, tid = threadIdx.x;
    const float* s = S + (size_t)row * lds;
    half_t* p = P + (size_t)row * ldp;
    float mx = -3.4e38f;
    for (int c = tid; c < Nv; c += 256) mx = fmaxf(mx, s[c]);
    red[tid] = mx; __syncthreads();
    for (int w = 128; w > 0; w >>= 1) {
        if (tid < w) red[tid] = fmaxf(red[tid], red[tid + w]);
        __syncthreads();
    }
    mx = red[0]; __syncthreads();
    float sum = 0.f;
    for (int c = tid; c < Nv; c += 256) sum += __expf(s[c] - mx);
    red[tid] = sum; __syncthreads();
    for (int w = 128; w > 0; w >>= 1) {
        if (tid < w) red[tid] += red[tid + w];
        __syncthreads();
    }
    float inv = 1.f / red[0];
    for (int c = tid; c < Nv; c += 256) p[c] = (half_t)(__expf(s[c] - mx) * inv);
    for (int c = Nv + tid; c < Np; c += 256) p[c] = (half_t)0.f;  // zero K-pad
}

__global__ __launch_bounds__(256)
void l2norm_f16_k(const float* __restrict__ in, half_t* __restrict__ out, int D)
{
    __shared__ float red[256];
    int row = blockIdx.x, tid = threadIdx.x;
    const float* x = in + (size_t)row * D;
    float ss = 0.f;
    for (int c = tid; c < D; c += 256) { float v = x[c]; ss += v * v; }
    red[tid] = ss; __syncthreads();
    for (int w = 128; w > 0; w >>= 1) {
        if (tid < w) red[tid] += red[tid + w];
        __syncthreads();
    }
    float inv = 1.f / (sqrtf(red[0]) + 1e-6f);
    for (int c = tid; c < D; c += 256) out[(size_t)row * D + c] = (half_t)(x[c] * inv);
}

__global__ void f32_to_f16_k(const float* __restrict__ in, half_t* __restrict__ out,
                             size_t n)
{
    for (size_t i = (size_t)blockIdx.x * blockDim.x + threadIdx.x; i < n;
         i += (size_t)gridDim.x * blockDim.x)
        out[i] = (half_t)in[i];
}

// im2col for 16x16 patchify: K layout (c, iy, ix) matches flattened conv weight
__global__ void im2col_k(const float* __restrict__ px, half_t* __restrict__ out, int n)
{
    for (int i = blockIdx.x * blockDim.x + threadIdx.x; i < n;
         i += gridDim.x * blockDim.x) {
        int p = i / 768, j = i % 768;
        int c = j >> 8, rem = j & 255, iy = rem >> 4, ix = rem & 15;
        int py = p / 60, pxx = p % 60;
        out[i] = (half_t)px[(size_t)c * 960 * 960 + (size_t)(py * 16 + iy) * 960
                            + (pxx * 16 + ix)];
    }
}

__global__ void embed_k(const float* __restrict__ conv, const float* __restrict__ cls,
                        const float* __restrict__ pos, float* __restrict__ h, int n)
{
    for (int i = blockIdx.x * blockDim.x + threadIdx.x; i < n;
         i += gridDim.x * blockDim.x) {
        int row = i / 768, c = i % 768;
        float v = (row == 0) ? cls[c] : conv[(size_t)(row - 1) * 768 + c];
        h[i] = v + pos[i];
    }
}

__global__ void featsmul_k(const float* __restrict__ img, float* __restrict__ out, int n)
{
    for (int i = blockIdx.x * blockDim.x + threadIdx.x; i < n;
         i += gridDim.x * blockDim.x) {
        int c = i % 768;
        out[i] = img[768 + i] * img[c];   // img[1+p]*img[0]
    }
}

__global__ __launch_bounds__(256)
void shiftscale_k(const float* __restrict__ feats,
                  const float* __restrict__ sw, const float* __restrict__ sb,
                  const float* __restrict__ cw, const float* __restrict__ cb,
                  float* __restrict__ out /* [M][2] */)
{
    __shared__ float r1[256], r2[256];
    int row = blockIdx.x, tid = threadIdx.x;
    const float* x = feats + (size_t)row * 768;
    float a = 0.f, b = 0.f;
    for (int c = tid; c < 768; c += 256) { float v = x[c]; a += v * sw[c]; b += v * cw[c]; }
    r1[tid] = a; r2[tid] = b; __syncthreads();
    for (int w = 128; w > 0; w >>= 1) {
        if (tid < w) { r1[tid] += r1[tid + w]; r2[tid] += r2[tid + w]; }
        __syncthreads();
    }
    if (tid == 0) {
        float sh = r1[0] + sb[0];
        float z  = r2[0] + cb[0];
        float sc = (z > 0.f ? z : (__expf(z) - 1.f)) + 1.f;   // elu(z)+1
        out[row * 2 + 0] = sh;
        out[row * 2 + 1] = sc;
    }
}

__global__ __launch_bounds__(256)
void boxfinal_k(const half_t* __restrict__ y, const float* __restrict__ w,
                const float* __restrict__ b, const float* __restrict__ box_bias,
                float* __restrict__ out)
{
    __shared__ float red[4][256];
    int row = blockIdx.x, tid = threadIdx.x;
    float acc[4] = {0.f, 0.f, 0.f, 0.f};
    for (int c = tid; c < 768; c += 256) {
        float v = (float)y[(size_t)row * 768 + c];
        for (int n = 0; n < 4; ++n) acc[n] += v * w[n * 768 + c];
    }
    for (int n = 0; n < 4; ++n) red[n][tid] = acc[n];
    __syncthreads();
    for (int w2 = 128; w2 > 0; w2 >>= 1) {
        if (tid < w2)
            for (int n = 0; n < 4; ++n) red[n][tid] += red[n][tid + w2];
        __syncthreads();
    }
    if (tid < 4) {
        float z = red[tid][0] + b[tid] + box_bias[row * 4 + tid];
        out[row * 4 + tid] = 1.f / (1.f + __expf(-z));
    }
}

__global__ void logits_epi_k(float* __restrict__ logits, const float* __restrict__ ss,
                             const unsigned char* __restrict__ mask, int n)
{
    for (int i = blockIdx.x * blockDim.x + threadIdx.x; i < n;
         i += gridDim.x * blockDim.x) {
        int m = i / 1203, k = i % 1203;
        float l = (logits[i] + ss[m * 2]) * ss[m * 2 + 1];
        if (!mask[k]) l = -1e6f;
        logits[i] = l;
    }
}

// ===========================================================================
// Host orchestration
// ===========================================================================
extern "C" void kernel_launch(void* const* d_in, const int* in_sizes, int n_in,
                              void* d_out, int out_size, void* d_ws, size_t ws_size,
                              hipStream_t stream)
{
    (void)in_sizes; (void)n_in; (void)out_size; (void)ws_size;

    const int S = 3601, PN = 3600, HID = 768, L = 12, HEADS = 12, HD = 64;
    const int INTER = 3072, QD = 512, NQ = 1203;
    const int SP = 3616;  // score/prob K-pad (multiple of 32)

    const float* pixel      = (const float*)d_in[0];
    const float* query_emb  = (const float*)d_in[1];
    const unsigned char* query_mask = (const unsigned char*)d_in[2];
    const float* patch_w    = (const float*)d_in[3];
    const float* class_emb  = (const float*)d_in[4];
    const float* pos_emb    = (const float*)d_in[5];
    const float* pre_ln_w   = (const float*)d_in[6];
    const float* pre_ln_b   = (const float*)d_in[7];
    const float* post_ln_w  = (const float*)d_in[8];
    const float* post_ln_b  = (const float*)d_in[9];
    const float* final_ln_w = (const float*)d_in[10];
    const float* final_ln_b = (const float*)d_in[11];
    const float* ln1_w = (const float*)d_in[12];
    const float* ln1_b = (const float*)d_in[13];
    const float* q_w   = (const float*)d_in[14];
    const float* q_b   = (const float*)d_in[15];
    const float* k_w   = (const float*)d_in[16];
    const float* k_b   = (const float*)d_in[17];
    const float* v_w   = (const float*)d_in[18];
    const float* v_b   = (const float*)d_in[19];
    const float* o_w   = (const float*)d_in[20];
    const float* o_b   = (const float*)d_in[21];
    const float* ln2_w = (const float*)d_in[22];
    const float* ln2_b = (const float*)d_in[23];
    const float* fc1_w = (const float*)d_in[24];
    const float* fc1_b = (const float*)d_in[25];
    const float* fc2_w = (const float*)d_in[26];
    const float* fc2_b = (const float*)d_in[27];
    const float* cls_w = (const float*)d_in[28];
    const float* cls_b = (const float*)d_in[29];
    const float* shift_w = (const float*)d_in[30];
    const float* shift_b = (const float*)d_in[31];
    const float* scale_w = (const float*)d_in[32];
    const float* scale_b = (const float*)d_in[33];
    const float* box_w0 = (const float*)d_in[34];
    const float* box_b0 = (const float*)d_in[35];
    const float* box_w1 = (const float*)d_in[36];
    const float* box_b1 = (const float*)d_in[37];
    const float* box_w2 = (const float*)d_in[38];
    const float* box_b2 = (const float*)d_in[39];
    const float* box_bias = (const float*)d_in[40];

    // bump allocator over workspace
    char* base = (char*)d_ws;
    size_t off = 0;
    auto allocB = [&](size_t bytes) -> void* {
        void* p = base + off;
        off = (off + bytes + 255) & ~(size_t)255;
        return p;
    };
    auto allocH = [&](size_t n) -> half_t* { return (half_t*)allocB(n * sizeof(half_t)); };
    auto allocF = [&](size_t n) -> float*  { return (float*)allocB(n * sizeof(float)); };

    // f16 weights
    half_t* patchW16 = allocH((size_t)HID * HID);
    half_t* qW16   = allocH((size_t)L * HID * HID);
    half_t* kW16   = allocH((size_t)L * HID * HID);
    half_t* vW16   = allocH((size_t)L * HID * HID);
    half_t* oW16   = allocH((size_t)L * HID * HID);
    half_t* fc1W16 = allocH((size_t)L * INTER * HID);
    half_t* fc2W16 = allocH((size_t)L * HID * INTER);
    half_t* clsW16 = allocH((size_t)QD * HID);
    half_t* b0W16  = allocH((size_t)HID * HID);
    half_t* b1W16  = allocH((size_t)HID * HID);

    // activations
    half_t* patches16 = allocH((size_t)PN * HID);
    float*  conv32    = allocF((size_t)PN * HID);
    float*  h32       = allocF((size_t)S * HID);
    half_t* x16       = allocH((size_t)S * HID);
    half_t* q16       = allocH((size_t)S * HID);
    half_t* k16       = allocH((size_t)S * HID);
    half_t* v16       = allocH((size_t)S * HID);
    half_t* attn16    = allocH((size_t)S * HID);
    half_t* y16       = allocH((size_t)S * HID);
    half_t* m16       = allocH((size_t)S * INTER);
    float*  S32       = allocF((size_t)S * SP);
    half_t* P16       = allocH((size_t)S * SP);

    // post-transformer buffers reuse transformer scratch
    float*  img32   = conv32;
    float*  feats32 = S32;
    half_t* feats16 = x16;
    float*  ic32    = (float*)P16;
    half_t* icn16   = q16;
    half_t* qn16    = k16;
    float*  ssbuf   = (float*)attn16;
    half_t* boxy1   = v16;
    half_t* boxy2   = y16;

    auto cvt = [&](const float* src, half_t* dst, size_t n) {
        int blocks = (int)((n + 255) / 256);
        if (blocks > 65535) blocks = 65535;
        f32_to_f16_k<<<blocks, 256, 0, stream>>>(src, dst, n);
    };
    auto gemm = [&](const half_t* A, int lda, const half_t* B, int ldb, int bKN,
                    const float* bias, const float* res, int ldr,
                    float* oF, half_t* oH, int ldc,
                    int M, int N, int K, int Kv, float alpha, int act) {
        dim3 g((N + BN - 1) / BN, (M + BM - 1) / BM);
        wmma_gemm_f16<<<g, dim3(256), 0, stream>>>(A, lda, B, ldb, bKN, bias, res, ldr,
                                                   oF, oH, ldc, M, N, K, Kv, alpha, act);
    };

    // ---- weight conversion (f32 -> f16), once per call ----
    cvt(patch_w, patchW16, (size_t)HID * HID);
    cvt(q_w,   qW16,   (size_t)L * HID * HID);
    cvt(k_w,   kW16,   (size_t)L * HID * HID);
    cvt(v_w,   vW16,   (size_t)L * HID * HID);
    cvt(o_w,   oW16,   (size_t)L * HID * HID);
    cvt(fc1_w, fc1W16, (size_t)L * INTER * HID);
    cvt(fc2_w, fc2W16, (size_t)L * HID * INTER);
    cvt(cls_w, clsW16, (size_t)QD * HID);
    cvt(box_w0, b0W16, (size_t)HID * HID);
    cvt(box_w1, b1W16, (size_t)HID * HID);

    // ---- embeddings: im2col + conv GEMM + class/pos add + pre-LN ----
    {
        int n = PN * HID;
        im2col_k<<<(n + 255) / 256 > 65535 ? 65535 : (n + 255) / 256, 256, 0, stream>>>(
            pixel, patches16, n);
        gemm(patches16, HID, patchW16, HID, 0, nullptr, nullptr, 0,
             conv32, nullptr, HID, PN, HID, HID, HID, 1.f, 0);
        int ne = S * HID;
        embed_k<<<(ne + 255) / 256 > 65535 ? 65535 : (ne + 255) / 256, 256, 0, stream>>>(
            conv32, class_emb, pos_emb, h32, ne);
        layernorm_k<<<S, 256, 0, stream>>>(h32, HID, pre_ln_w, pre_ln_b,
                                           h32, nullptr, HID, HID);
    }

    // ---- transformer layers ----
    for (int l = 0; l < L; ++l) {
        const size_t wo  = (size_t)l * HID * HID;
        const size_t wf1 = (size_t)l * INTER * HID;

        layernorm_k<<<S, 256, 0, stream>>>(h32, HID, ln1_w + l * HID, ln1_b + l * HID,
                                           nullptr, x16, HID, HID);
        gemm(x16, HID, qW16 + wo, HID, 0, q_b + l * HID, nullptr, 0,
             nullptr, q16, HID, S, HID, HID, HID, 0.125f, 0);   // q * HEAD_DIM^-0.5
        gemm(x16, HID, kW16 + wo, HID, 0, k_b + l * HID, nullptr, 0,
             nullptr, k16, HID, S, HID, HID, HID, 1.f, 0);
        gemm(x16, HID, vW16 + wo, HID, 0, v_b + l * HID, nullptr, 0,
             nullptr, v16, HID, S, HID, HID, HID, 1.f, 0);

        for (int hd = 0; hd < HEADS; ++hd) {
            // S = Q_h K_h^T   [S x S], K = 64
            gemm(q16 + hd * HD, HID, k16 + hd * HD, HID, 0, nullptr, nullptr, 0,
                 S32, nullptr, SP, S, S, HD, HD, 1.f, 0);
            softmax_f16_k<<<S, 256, 0, stream>>>(S32, SP, P16, SP, S, SP);
            // O_h = P V_h     [S x 64], K = 3616 (zero-padded), B is [K,N]
            gemm(P16, SP, v16 + hd * HD, HID, 1, nullptr, nullptr, 0,
                 nullptr, attn16 + hd * HD, HID, S, HD, SP, S, 1.f, 0);
        }

        // h = h + attn @ o_w^T + o_b
        gemm(attn16, HID, oW16 + wo, HID, 0, o_b + l * HID, h32, HID,
             h32, nullptr, HID, S, HID, HID, HID, 1.f, 0);

        layernorm_k<<<S, 256, 0, stream>>>(h32, HID, ln2_w + l * HID, ln2_b + l * HID,
                                           nullptr, y16, HID, HID);
        gemm(y16, HID, fc1W16 + wf1, HID, 0, fc1_b + (size_t)l * INTER, nullptr, 0,
             nullptr, m16, INTER, S, INTER, HID, HID, 1.f, 1);  // quick_gelu
        gemm(m16, INTER, fc2W16 + wf1, INTER, 0, fc2_b + l * HID, h32, HID,
             h32, nullptr, HID, S, HID, INTER, INTER, 1.f, 0);
    }

    // ---- post-LN, class-token merge, final LN ----
    layernorm_k<<<S, 256, 0, stream>>>(h32, HID, post_ln_w, post_ln_b,
                                       img32, nullptr, HID, HID);
    {
        int n = PN * HID;
        featsmul_k<<<(n + 255) / 256 > 65535 ? 65535 : (n + 255) / 256, 256, 0, stream>>>(
            img32, feats32, n);
    }
    layernorm_k<<<PN, 256, 0, stream>>>(feats32, HID, final_ln_w, final_ln_b,
                                        feats32, feats16, HID, HID);

    // ---- class prediction head ----
    gemm(feats16, HID, clsW16, HID, 0, cls_b, nullptr, 0,
         ic32, nullptr, QD, PN, QD, HID, HID, 1.f, 0);
    l2norm_f16_k<<<PN, 256, 0, stream>>>(ic32, icn16, QD);
    l2norm_f16_k<<<NQ, 256, 0, stream>>>(query_emb, qn16, QD);

    float* logits = (float*)d_out;                          // [3600 x 1203]
    float* boxes  = (float*)d_out + (size_t)PN * NQ;        // [3600 x 4]
    gemm(icn16, QD, qn16, QD, 0, nullptr, nullptr, 0,
         logits, nullptr, NQ, PN, NQ, QD, QD, 1.f, 0);
    shiftscale_k<<<PN, 256, 0, stream>>>(feats32, shift_w, shift_b, scale_w, scale_b,
                                         ssbuf);
    {
        int n = PN * NQ;
        logits_epi_k<<<(n + 255) / 256 > 65535 ? 65535 : (n + 255) / 256, 256, 0,
                      stream>>>(logits, ssbuf, query_mask, n);
    }

    // ---- box prediction head ----
    gemm(feats16, HID, b0W16, HID, 0, box_b0, nullptr, 0,
         nullptr, boxy1, HID, PN, HID, HID, HID, 1.f, 2);       // exact gelu
    gemm(boxy1, HID, b1W16, HID, 0, box_b1, nullptr, 0,
         nullptr, boxy2, HID, PN, HID, HID, HID, 1.f, 2);
    boxfinal_k<<<PN, 256, 0, stream>>>(boxy2, box_w2, box_b2, box_bias, boxes);
}
